// LSTM_41729902248037
// MI455X (gfx1250) — matrix operand order repaired and checked by reading, not compiled, not run on hardware
//
#include <hip/hip_runtime.h>
#include <hip/hip_bf16.h>

typedef __attribute__((ext_vector_type(16))) __bf16 v16bf;
typedef __attribute__((ext_vector_type(8)))  float  v8f;

#define T_STEPS 256
#define BATCH   64
#define IDIM    1024
#define HDIM    1024
#define G4H     4096
#define SCAN_BLOCKS 64

union ABTile {
    v16bf v;
    uint4 u[2];
};

__device__ __forceinline__ unsigned short f2bf(float f) {
    unsigned int u = __float_as_uint(f);
    unsigned int r = (u + 0x7FFFu + ((u >> 16) & 1u)) >> 16;
    return (unsigned short)r;
}

__device__ __forceinline__ float sigmoidf(float x) {
    return 1.0f / (1.0f + __expf(-x));
}

// Read a 16x32 A tile from an LDS slab laid out [row][32k] (64B rows).
// ISA 16-bit A layout: lanes0-15 K[0..7],K[16..23]; lanes16-31 K[8..15],K[24..31]
__device__ __forceinline__ void lds_read_a(ABTile& a, const unsigned short* lds,
                                           int row, int hi) {
    const unsigned short* p = lds + row * 32 + hi * 8;
    a.u[0] = *(const uint4*)(p);
    a.u[1] = *(const uint4*)(p + 16);
}

// Read a 32x16 B tile (col-packed rows) from an LDS slab [row][32k].
// ISA 16-bit B layout: lanes0-15 K[0..7],K[16..23]; lanes16-31 K[8..15],K[24..31]
__device__ __forceinline__ void lds_read_b(ABTile& b, const unsigned short* lds,
                                           int row, int hi) {
    const unsigned short* p = lds + row * 32 + hi * 8;
    b.u[0] = *(const uint4*)(p);
    b.u[1] = *(const uint4*)(p + 16);
}

// ---------------- prologue: convert / pack ----------------

__global__ void __launch_bounds__(256)
cvt_f32_bf16(const float* __restrict__ src, unsigned short* __restrict__ dst, int n) {
    int i = blockIdx.x * blockDim.x + threadIdx.x;
    if (i < n) dst[i] = f2bf(src[i]);
}

// src [rows, cols] fp32 row-major  ->  dst [cols, rows] bf16 row-major
__global__ void __launch_bounds__(256)
pack_T_bf16(const float* __restrict__ src, unsigned short* __restrict__ dst,
            int rows, int cols) {
    int i = blockIdx.x * blockDim.x + threadIdx.x;
    if (i >= rows * cols) return;
    int r = i / cols, c = i % cols;
    dst[c * rows + r] = f2bf(src[i]);
}

__global__ void __launch_bounds__(256)
init_state(const float* __restrict__ h0, unsigned short* __restrict__ hbf,
           unsigned int* __restrict__ barrier_ctr) {
    int i = blockIdx.x * blockDim.x + threadIdx.x;
    if (i < BATCH * HDIM) hbf[i] = f2bf(h0[i]);
    if (i == 0) barrier_ctr[0] = 0u;
}

// ---------------- big GEMM: Wx = x @ kernel ----------------
// Block tile 128x128 (8 waves = 4 m-groups x 2 n-groups, each wave 32x64).
// A/B K-slabs staged in LDS (double buffered), shared across the block.
__global__ void __launch_bounds__(256)
wx_gemm(const unsigned short* __restrict__ Abf,   // [16384,1024] bf16
        const unsigned short* __restrict__ Bt,    // [4096,1024] bf16 (W^T)
        float* __restrict__ C) {                  // [16384,4096] fp32
    __shared__ unsigned short aS[2][128 * 32];    // 8 KB per buffer
    __shared__ unsigned short bS[2][128 * 32];

    const int tid   = threadIdx.x;
    const int lane  = tid & 31;
    const int wave  = tid >> 5;             // 0..7
    const int mw    = wave & 3;             // m wave-group (rows mw*32)
    const int nw    = wave >> 2;            // n wave-group (cols nw*64)
    const int mBase = (blockIdx.x >> 5) * 128;   // 128 m-blocks
    const int nBase = (blockIdx.x & 31) * 128;   // 32 n-blocks
    const int l15   = lane & 15;
    const int hi    = lane >> 4;

    v8f acc[2][4];
    #pragma unroll
    for (int m = 0; m < 2; ++m)
        #pragma unroll
        for (int g = 0; g < 4; ++g)
            #pragma unroll
            for (int i = 0; i < 8; ++i) acc[m][g][i] = 0.0f;

    // slab chunks: 128 rows x 64B = 512 chunks of 16B; 2 per thread
    const int c0id = tid, c1id = tid + 256;
    const unsigned short* gA0 = Abf + (mBase + (c0id >> 2)) * IDIM + (c0id & 3) * 8;
    const unsigned short* gA1 = Abf + (mBase + (c1id >> 2)) * IDIM + (c1id & 3) * 8;
    const unsigned short* gB0 = Bt + (nBase + (c0id >> 2)) * IDIM + (c0id & 3) * 8;
    const unsigned short* gB1 = Bt + (nBase + (c1id >> 2)) * IDIM + (c1id & 3) * 8;

    uint4 ra0, ra1, rb0, rb1;
    ra0 = *(const uint4*)(gA0); ra1 = *(const uint4*)(gA1);
    rb0 = *(const uint4*)(gB0); rb1 = *(const uint4*)(gB1);
    *(uint4*)(&aS[0][c0id * 8]) = ra0; *(uint4*)(&aS[0][c1id * 8]) = ra1;
    *(uint4*)(&bS[0][c0id * 8]) = rb0; *(uint4*)(&bS[0][c1id * 8]) = rb1;
    __syncthreads();

    int cur = 0;
    for (int k0 = 0; k0 < IDIM; k0 += 32) {
        const int nxt = k0 + 32;
        if (nxt < IDIM) {   // prefetch next slab global -> regs
            ra0 = *(const uint4*)(gA0 + nxt); ra1 = *(const uint4*)(gA1 + nxt);
            rb0 = *(const uint4*)(gB0 + nxt); rb1 = *(const uint4*)(gB1 + nxt);
        }
        // compute from current LDS slab
        ABTile at[2], bt[4];
        lds_read_a(at[0], aS[cur], mw * 32 + l15, hi);
        lds_read_a(at[1], aS[cur], mw * 32 + 16 + l15, hi);
        #pragma unroll
        for (int g = 0; g < 4; ++g)
            lds_read_b(bt[g], bS[cur], nw * 64 + g * 16 + l15, hi);
        #pragma unroll
        for (int g = 0; g < 4; ++g) {
            acc[0][g] = __builtin_amdgcn_wmma_f32_16x16x32_bf16(
                false, at[0].v, false, bt[g].v, (short)0, acc[0][g], false, false);
            acc[1][g] = __builtin_amdgcn_wmma_f32_16x16x32_bf16(
                false, at[1].v, false, bt[g].v, (short)0, acc[1][g], false, false);
        }
        if (nxt < IDIM) {   // stage next slab into the other LDS buffer
            *(uint4*)(&aS[1 - cur][c0id * 8]) = ra0;
            *(uint4*)(&aS[1 - cur][c1id * 8]) = ra1;
            *(uint4*)(&bS[1 - cur][c0id * 8]) = rb0;
            *(uint4*)(&bS[1 - cur][c1id * 8]) = rb1;
        }
        __syncthreads();
        cur ^= 1;
    }

    #pragma unroll
    for (int m = 0; m < 2; ++m)
        #pragma unroll
        for (int g = 0; g < 4; ++g) {
            int col = nBase + nw * 64 + g * 16 + l15;
            #pragma unroll
            for (int i = 0; i < 8; ++i) {
                int row = mBase + mw * 32 + m * 16 + hi * 8 + i;
                C[row * G4H + col] = acc[m][g][i];
            }
        }
}

// ---------------- persistent fused scan ----------------
// 64 blocks x 128 threads; block = one 16-wide hidden-column tile, its 4 waves
// are the 4 batch row-blocks. Rt/h K-slabs staged once in LDS and shared by
// all 4 waves -> Rt is read from L2 exactly once per timestep grid-wide.
// Cell state c lives in registers for the whole scan. Grid sync via atomics.
__global__ void __launch_bounds__(128)
lstm_scan(const unsigned short* __restrict__ Rt,   // [4096,1024] bf16 (R^T)
          const float* __restrict__ Wx,            // [T*B, 4096]
          const float* __restrict__ bias,          // [4096]
          const float* __restrict__ c0,            // [64,1024]
          unsigned short* __restrict__ hbf0,       // ping
          unsigned short* __restrict__ hbf1,       // pong
          float* __restrict__ hs,                  // [T,64,1024] (d_out)
          float* __restrict__ hn_out,              // [64,1024]
          float* __restrict__ cn_out,              // [64,1024]
          unsigned int* __restrict__ barrier_ctr) {
    __shared__ unsigned short aS[2][64 * 32];      // h slab: 64 batch x 32 k
    __shared__ unsigned short bS[2][64 * 32];      // Rt slab: (4g x 16 col) x 32 k

    const int tid   = threadIdx.x;                 // 0..127
    const int lane  = tid & 31;
    const int wave  = tid >> 5;                    // 0..3 = batch row-block
    const int nb    = blockIdx.x;                  // 0..63 hidden col tile
    const int rBase = wave * 16;
    const int nBase = nb * 16;
    const int l15   = lane & 15;
    const int hi    = lane >> 4;
    const int col   = nBase + l15;

    // per-lane constants in registers for the whole scan
    float bg[4];
    #pragma unroll
    for (int g = 0; g < 4; ++g) bg[g] = bias[g * HDIM + col];

    v8f cstate;
    #pragma unroll
    for (int i = 0; i < 8; ++i)
        cstate[i] = c0[(rBase + hi * 8 + i) * HDIM + col];

    // slab chunks: 64 rows x 64B = 256 chunks of 16B; 2 per thread
    const int c0id = tid, c1id = tid + 128;
    // B slab sources: slab row = g*16 + j  ->  Rt row (g*HDIM + nBase + j)
    const unsigned short* gB0 =
        Rt + (((c0id >> 6) * HDIM) + nBase + ((c0id >> 2) & 15)) * HDIM + (c0id & 3) * 8;
    const unsigned short* gB1 =
        Rt + (((c1id >> 6) * HDIM) + nBase + ((c1id >> 2) & 15)) * HDIM + (c1id & 3) * 8;

    for (int t = 0; t < T_STEPS; ++t) {
        const unsigned short* hin  = (t & 1) ? hbf1 : hbf0;
        unsigned short*       hout = (t & 1) ? hbf0 : hbf1;

        // seed accumulators with Wx[t] + bias (WMMA C layout)
        v8f acc[4];
        #pragma unroll
        for (int g = 0; g < 4; ++g) {
            v8f c;
            #pragma unroll
            for (int i = 0; i < 8; ++i) {
                int row = rBase + hi * 8 + i;
                c[i] = Wx[(t * BATCH + row) * G4H + g * HDIM + col] + bg[g];
            }
            acc[g] = c;
        }

        const unsigned short* gA0 = hin + (c0id >> 2) * HDIM + (c0id & 3) * 8;
        const unsigned short* gA1 = hin + (c1id >> 2) * HDIM + (c1id & 3) * 8;

        uint4 ra0, ra1, rb0, rb1;
        ra0 = *(const uint4*)(gA0); ra1 = *(const uint4*)(gA1);
        rb0 = *(const uint4*)(gB0); rb1 = *(const uint4*)(gB1);
        *(uint4*)(&aS[0][c0id * 8]) = ra0; *(uint4*)(&aS[0][c1id * 8]) = ra1;
        *(uint4*)(&bS[0][c0id * 8]) = rb0; *(uint4*)(&bS[0][c1id * 8]) = rb1;
        __syncthreads();

        int cur = 0;
        for (int k0 = 0; k0 < HDIM; k0 += 32) {
            const int nxt = k0 + 32;
            if (nxt < HDIM) {
                ra0 = *(const uint4*)(gA0 + nxt); ra1 = *(const uint4*)(gA1 + nxt);
                rb0 = *(const uint4*)(gB0 + nxt); rb1 = *(const uint4*)(gB1 + nxt);
            }
            ABTile at, bt[4];
            lds_read_a(at, aS[cur], rBase + l15, hi);
            #pragma unroll
            for (int g = 0; g < 4; ++g)
                lds_read_b(bt[g], bS[cur], g * 16 + l15, hi);
            #pragma unroll
            for (int g = 0; g < 4; ++g)
                acc[g] = __builtin_amdgcn_wmma_f32_16x16x32_bf16(
                    false, at.v, false, bt[g].v, (short)0, acc[g], false, false);
            if (nxt < HDIM) {
                *(uint4*)(&aS[1 - cur][c0id * 8]) = ra0;
                *(uint4*)(&aS[1 - cur][c1id * 8]) = ra1;
                *(uint4*)(&bS[1 - cur][c0id * 8]) = rb0;
                *(uint4*)(&bS[1 - cur][c1id * 8]) = rb1;
            }
            __syncthreads();
            cur ^= 1;
        }

        // gates + state update in the WMMA C/D register layout
        #pragma unroll
        for (int i = 0; i < 8; ++i) {
            int row = rBase + hi * 8 + i;
            float iv = sigmoidf(acc[0][i]);
            float gv = tanhf(acc[1][i]);
            float fv = sigmoidf(acc[2][i]);
            float ov = sigmoidf(acc[3][i]);
            float cn = fv * cstate[i] + iv * gv;
            float hn = ov * tanhf(cn);
            cstate[i] = cn;
            int idx = row * HDIM + col;
            hs[(size_t)t * (BATCH * HDIM) + idx] = hn;
            hout[idx] = f2bf(hn);
            if (t == T_STEPS - 1) {
                hn_out[idx] = hn;
                cn_out[idx] = cn;
            }
        }

        // device-wide barrier: monotonic counter, release/acquire, agent scope
        __syncthreads();
        if (tid == 0) {
            __hip_atomic_fetch_add(barrier_ctr, 1u, __ATOMIC_RELEASE,
                                   __HIP_MEMORY_SCOPE_AGENT);
            unsigned int target = (unsigned int)SCAN_BLOCKS * (unsigned int)(t + 1);
            while (__hip_atomic_load(barrier_ctr, __ATOMIC_ACQUIRE,
                                     __HIP_MEMORY_SCOPE_AGENT) < target) {
                __builtin_amdgcn_s_sleep(1);
            }
        }
        __syncthreads();
    }
}

// ---------------- launcher ----------------

extern "C" void kernel_launch(void* const* d_in, const int* in_sizes, int n_in,
                              void* d_out, int out_size, void* d_ws, size_t ws_size,
                              hipStream_t stream) {
    const float* x    = (const float*)d_in[0];  // [256,64,1024]
    const float* h0   = (const float*)d_in[1];  // [64,1024]
    const float* c0   = (const float*)d_in[2];  // [64,1024]
    const float* Wk   = (const float*)d_in[3];  // [1024,4096]
    const float* Rk   = (const float*)d_in[4];  // [1024,4096]
    const float* bias = (const float*)d_in[5];  // [4096]

    char* ws = (char*)d_ws;
    size_t off = 0;
    unsigned short* x_bf = (unsigned short*)(ws + off); off += (size_t)T_STEPS * BATCH * IDIM * 2;
    unsigned short* Wt   = (unsigned short*)(ws + off); off += (size_t)G4H * IDIM * 2;
    unsigned short* Rt   = (unsigned short*)(ws + off); off += (size_t)G4H * HDIM * 2;
    float*          Wx   = (float*)(ws + off);          off += (size_t)T_STEPS * BATCH * G4H * 4;
    unsigned short* hbf0 = (unsigned short*)(ws + off); off += (size_t)BATCH * HDIM * 2;
    unsigned short* hbf1 = (unsigned short*)(ws + off); off += (size_t)BATCH * HDIM * 2;
    unsigned int*   bctr = (unsigned int*)(ws + off);

    float* hs = (float*)d_out;                          // [256,64,1024]
    float* hn = hs + (size_t)T_STEPS * BATCH * HDIM;    // [64,1024]
    float* cn = hn + (size_t)BATCH * HDIM;              // [64,1024]

    const int nX = T_STEPS * BATCH * IDIM;
    cvt_f32_bf16<<<nX / 256, 256, 0, stream>>>(x, x_bf, nX);
    pack_T_bf16<<<(IDIM * G4H) / 256, 256, 0, stream>>>(Wk, Wt, IDIM, G4H);
    pack_T_bf16<<<(HDIM * G4H) / 256, 256, 0, stream>>>(Rk, Rt, HDIM, G4H);
    init_state<<<(BATCH * HDIM) / 256, 256, 0, stream>>>(h0, hbf0, bctr);

    // Wx GEMM: 128 m-blocks x 32 n-blocks, block tile 128x128
    wx_gemm<<<4096, 256, 0, stream>>>(x_bf, Wt, Wx);

    // One persistent launch for the entire 256-step scan
    lstm_scan<<<SCAN_BLOCKS, 128, 0, stream>>>(Rt, Wx, bias, c0, hbf0, hbf1,
                                               hs, hn, cn, bctr);
}
